// Entropic_Wasserstein_26706106647244
// MI455X (gfx1250) — matrix operand order repaired and checked by reading, not compiled, not run on hardware
//
#include <hip/hip_runtime.h>
#include <hip/hip_bf16.h>
#include <math.h>

// Entropic Wasserstein (Sinkhorn), N=M=8192, D=32, EPS=10, 50 iters.
// K = exp(-C/EPS) built ONCE with v_wmma_f32_16x16x32_f16, stored bf16
// (128 MB -> resident in the 192 MB L2). The 100 matvec passes then stream
// from L2 with b128 loads (16 bytes per vmem instruction).

#define NROWS 8192
#define NCOLS 8192
#define DDIM  32
#define EPSI  10.0f
#define THRESH 1e-5f
#define MAXIT 50

#define VCHUNKS 64          // row chunks for K^T matvec
#define VROWS   (NROWS / VCHUNKS)   // 128 rows per chunk

typedef __attribute__((ext_vector_type(16))) _Float16 v16h;
typedef __attribute__((ext_vector_type(8)))  float    v8f;

__device__ __forceinline__ float bflo(unsigned int u) {
    return __uint_as_float(u << 16);
}
__device__ __forceinline__ float bfhi(unsigned int u) {
    return __uint_as_float(u & 0xffff0000u);
}
__device__ __forceinline__ float bf2f(unsigned short u) {
    return __uint_as_float(((unsigned int)u) << 16);
}
__device__ __forceinline__ unsigned short f2bf(float f) {
    unsigned int u = __float_as_uint(f);
    u += 0x7fffu + ((u >> 16) & 1u);   // round-to-nearest-even
    return (unsigned short)(u >> 16);
}

// ---- row sums of squares: x2[i] = sum_d x[i][d]^2 (and y2) -----------------
__global__ void __launch_bounds__(256) rowsq_kernel(
    const float* __restrict__ x, const float* __restrict__ y,
    float* __restrict__ x2, float* __restrict__ y2)
{
    int t = blockIdx.x * 256 + threadIdx.x;             // 16384 threads
    const float* src = (t < NROWS) ? x : y;
    float*       dst = (t < NROWS) ? x2 : y2;
    int r = (t < NROWS) ? t : (t - NROWS);
    const float4* v = (const float4*)(src + (size_t)r * DDIM);
    float s = 0.0f;
#pragma unroll
    for (int c = 0; c < 8; ++c) {
        float4 f = v[c];
        s += f.x * f.x + f.y * f.y + f.z * f.z + f.w * f.w;
    }
    dst[r] = s;
}

// ---- reset per-call state ---------------------------------------------------
__global__ void sink_init(float* __restrict__ b, int* __restrict__ done)
{
    int t = blockIdx.x * 256 + threadIdx.x;
    if (t < NCOLS) b[t] = 1.0f;
    if (t == 0) *done = 0;
}

// ---- build K (bf16) via WMMA: S = x @ y^T, C = x2+y2-2S, K = exp(-C/EPS) ---
// grid (512, 64), block 256 (= 8 waves); each wave makes one 16x16 tile.
__global__ void __launch_bounds__(256) build_K(
    const float* __restrict__ x, const float* __restrict__ y,
    const float* __restrict__ x2, const float* __restrict__ y2,
    unsigned short* __restrict__ K)
{
    const int lane = threadIdx.x & 31;
    const int wave = threadIdx.x >> 5;
    const int half = lane >> 4;        // 0: lanes 0-15, 1: lanes 16-31
    const int l15  = lane & 15;
    const int tm = (blockIdx.y * 8 + wave) * 16;
    const int tn = blockIdx.x * 16;

    // A (16x32 f16): lane holds row tm+l15; half=0 -> K 0..15,
    //                                      half=1 -> K 16..31
    v16h av, bv;
    {
        const float* xp = x + (size_t)(tm + l15) * DDIM + (half ? 16 : 0);
#pragma unroll
        for (int t = 0; t < 16; ++t) av[t] = (_Float16)xp[t];
        // B (32x16 f16): lane holds col tn+l15; half=0 -> K 0..15, half=1 -> 16..31
        const float* yp = y + (size_t)(tn + l15) * DDIM + (half ? 16 : 0);
#pragma unroll
        for (int t = 0; t < 16; ++t) bv[t] = (_Float16)yp[t];
    }

    v8f c = {};
    c = __builtin_amdgcn_wmma_f32_16x16x32_f16(
            false, av, false, bv, (short)0, c, false, false);

    const int j = tn + l15;
    const float y2j = y2[j];
#pragma unroll
    for (int r = 0; r < 8; ++r) {
        const int i = tm + half * 8 + r;                 // C/D layout
        float cc = fmaxf(x2[i] + y2j - 2.0f * c[r], 0.0f);
        float kv = __expf(cc * (-1.0f / EPSI));
        K[(size_t)i * NCOLS + j] = f2bf(kv);
    }
}

// ---- u = K @ b ; a = p / u. One wave per row; b128 loads for K and b. ------
__global__ void __launch_bounds__(256) sink_u(
    const unsigned short* __restrict__ K, const float* __restrict__ p,
    const float* __restrict__ b, float* __restrict__ a,
    const int* __restrict__ done)
{
    if (*done) return;
    const int lane = threadIdx.x & 31;
    const int wave = threadIdx.x >> 5;
    const int i = blockIdx.x * 8 + wave;
    const unsigned short* row = K + (size_t)i * NCOLS;
    float acc = 0.0f;
    for (int j0 = lane * 8; j0 < NCOLS; j0 += 256) {
        uint4  pk = *(const uint4*)(row + j0);           // 8 bf16, 16B aligned
        float4 b0 = *(const float4*)(b + j0);
        float4 b1 = *(const float4*)(b + j0 + 4);
        acc = fmaf(bflo(pk.x), b0.x, acc);
        acc = fmaf(bfhi(pk.x), b0.y, acc);
        acc = fmaf(bflo(pk.y), b0.z, acc);
        acc = fmaf(bfhi(pk.y), b0.w, acc);
        acc = fmaf(bflo(pk.z), b1.x, acc);
        acc = fmaf(bfhi(pk.z), b1.y, acc);
        acc = fmaf(bflo(pk.w), b1.z, acc);
        acc = fmaf(bfhi(pk.w), b1.w, acc);
    }
#pragma unroll
    for (int off = 16; off > 0; off >>= 1) acc += __shfl_down(acc, off, 32);
    if (lane == 0) a[i] = p[i] / acc;
}

// ---- v = K^T @ a, stage 1: partial sums over VROWS-row chunks --------------
// grid (4, VCHUNKS), block 256. Each thread owns 8 adjacent columns:
// one b128 K load + one uniform a[i] load + 8 FMAs per row.
__global__ void __launch_bounds__(256) sink_v(
    const unsigned short* __restrict__ K, const float* __restrict__ a,
    float* __restrict__ vpart, const int* __restrict__ done)
{
    if (*done) return;
    const int j0 = blockIdx.x * 2048 + threadIdx.x * 8;
    const int i0 = blockIdx.y * VROWS;
    const unsigned short* ptr = K + (size_t)i0 * NCOLS + j0;
    float acc[8];
#pragma unroll
    for (int t = 0; t < 8; ++t) acc[t] = 0.0f;
#pragma unroll 4
    for (int i = 0; i < VROWS; ++i) {
        uint4 pk = *(const uint4*)ptr;
        ptr += NCOLS;
        const float ai = a[i0 + i];        // uniform -> scalar load
        acc[0] = fmaf(bflo(pk.x), ai, acc[0]);
        acc[1] = fmaf(bfhi(pk.x), ai, acc[1]);
        acc[2] = fmaf(bflo(pk.y), ai, acc[2]);
        acc[3] = fmaf(bfhi(pk.y), ai, acc[3]);
        acc[4] = fmaf(bflo(pk.z), ai, acc[4]);
        acc[5] = fmaf(bfhi(pk.z), ai, acc[5]);
        acc[6] = fmaf(bflo(pk.w), ai, acc[6]);
        acc[7] = fmaf(bfhi(pk.w), ai, acc[7]);
    }
    float* vp = vpart + (size_t)blockIdx.y * NCOLS + j0;
    *(float4*)(vp)     = make_float4(acc[0], acc[1], acc[2], acc[3]);
    *(float4*)(vp + 4) = make_float4(acc[4], acc[5], acc[6], acc[7]);
}

// ---- stage 2: reduce partials, b_new = q/v, err partials (deterministic) ---
__global__ void __launch_bounds__(256) sink_v2(
    const float* __restrict__ vpart, const float* __restrict__ q,
    float* __restrict__ b, float* __restrict__ errp,
    const int* __restrict__ done)
{
    if (*done) return;
    const int j = blockIdx.x * 256 + threadIdx.x;
    float v = 0.0f;
#pragma unroll 8
    for (int t = 0; t < VCHUNKS; ++t) v += vpart[(size_t)t * NCOLS + j];
    float bn = q[j] / v;
    float d  = bn - b[j];
    b[j] = bn;
    __shared__ float red[256];
    red[threadIdx.x] = d * d;
    __syncthreads();
    for (int s = 128; s > 0; s >>= 1) {
        if (threadIdx.x < s) red[threadIdx.x] += red[threadIdx.x + s];
        __syncthreads();
    }
    if (threadIdx.x == 0) errp[blockIdx.x] = red[0];
}

// ---- convergence check: freeze once ||b_new - b|| < THRESH -----------------
__global__ void sink_check(const float* __restrict__ errp, int* __restrict__ done)
{
    if (*done) return;
    float s = 0.0f;
    for (int t = 0; t < 32; ++t) s += errp[t];   // fixed order, deterministic
    if (sqrtf(s) < THRESH) *done = 1;
}

// ---- T = a_i K_ij b_j ; cost partials with C = -EPS*ln(K) ------------------
// grid 4096, block 256; each block covers 2 rows (16384 elements) in 8 groups
// of 8 adjacent elements per thread (groups never straddle a row boundary).
__global__ void __launch_bounds__(256) finalize_T(
    const unsigned short* __restrict__ K, const float* __restrict__ a,
    const float* __restrict__ b, float* __restrict__ T,
    float* __restrict__ costp)
{
    __shared__ float red[256];
    const size_t base = (size_t)blockIdx.x * 16384;
    float local = 0.0f;
#pragma unroll
    for (int g = 0; g < 8; ++g) {
        const size_t idx = base + (size_t)g * 2048 + (size_t)threadIdx.x * 8;
        const int i = (int)(idx >> 13);
        const int j = (int)(idx & 8191);
        const float ai = a[i];
        uint4  pk = *(const uint4*)(K + idx);
        float4 b0 = *(const float4*)(b + j);
        float4 b1 = *(const float4*)(b + j + 4);
        float kf[8] = { bflo(pk.x), bfhi(pk.x), bflo(pk.y), bfhi(pk.y),
                        bflo(pk.z), bfhi(pk.z), bflo(pk.w), bfhi(pk.w) };
        float bv[8] = { b0.x, b0.y, b0.z, b0.w, b1.x, b1.y, b1.z, b1.w };
        float tv[8];
#pragma unroll
        for (int e = 0; e < 8; ++e) {
            tv[e] = ai * kf[e] * bv[e];
            if (kf[e] > 0.0f) local = fmaf(tv[e], -EPSI * __logf(kf[e]), local);
        }
        *(float4*)(T + idx)     = make_float4(tv[0], tv[1], tv[2], tv[3]);
        *(float4*)(T + idx + 4) = make_float4(tv[4], tv[5], tv[6], tv[7]);
    }
    red[threadIdx.x] = local;
    __syncthreads();
    for (int s = 128; s > 0; s >>= 1) {
        if (threadIdx.x < s) red[threadIdx.x] += red[threadIdx.x + s];
        __syncthreads();
    }
    if (threadIdx.x == 0) costp[blockIdx.x] = red[0];
}

__global__ void __launch_bounds__(256) cost_reduce(
    const float* __restrict__ costp, float* __restrict__ out)
{
    __shared__ float red[256];
    float s = 0.0f;
    for (int t = threadIdx.x; t < 4096; t += 256) s += costp[t];
    red[threadIdx.x] = s;
    __syncthreads();
    for (int st = 128; st > 0; st >>= 1) {
        if (threadIdx.x < st) red[threadIdx.x] += red[threadIdx.x + st];
        __syncthreads();
    }
    if (threadIdx.x == 0) out[0] = red[0];
}

extern "C" void kernel_launch(void* const* d_in, const int* in_sizes, int n_in,
                              void* d_out, int out_size, void* d_ws, size_t ws_size,
                              hipStream_t stream) {
    const float* x = (const float*)d_in[0];   // [8192,32]
    const float* y = (const float*)d_in[1];   // [8192,32]
    const float* p = (const float*)d_in[2];   // [8192]
    const float* q = (const float*)d_in[3];   // [8192]
    float* out = (float*)d_out;               // [0]=w_cost, [1..]=T flat
    float* T   = out + 1;

    // workspace carve (~137 MB; K bf16 is L2-resident at 128 MB)
    size_t off = 0;
    char* wsb = (char*)d_ws;
    auto carve = [&](size_t bytes) {
        void* ptr = wsb + off;
        off += (bytes + 255) & ~(size_t)255;
        return ptr;
    };
    unsigned short* K  = (unsigned short*)carve((size_t)NROWS * NCOLS * 2);
    float* x2    = (float*)carve(NROWS * 4);
    float* y2    = (float*)carve(NCOLS * 4);
    float* a     = (float*)carve(NROWS * 4);
    float* b     = (float*)carve(NCOLS * 4);
    float* vpart = (float*)carve((size_t)VCHUNKS * NCOLS * 4);
    float* errp  = (float*)carve(32 * 4);
    float* costp = (float*)carve(4096 * 4);
    int*   done  = (int*)carve(256);

    rowsq_kernel<<<64, 256, 0, stream>>>(x, y, x2, y2);
    sink_init<<<32, 256, 0, stream>>>(b, done);
    build_K<<<dim3(512, 64), 256, 0, stream>>>(x, y, x2, y2, K);

    for (int it = 0; it < MAXIT; ++it) {
        sink_u<<<1024, 256, 0, stream>>>(K, p, b, a, done);
        sink_v<<<dim3(4, VCHUNKS), 256, 0, stream>>>(K, a, vpart, done);
        sink_v2<<<32, 256, 0, stream>>>(vpart, q, b, errp, done);
        sink_check<<<1, 1, 0, stream>>>(errp, done);
    }

    finalize_T<<<4096, 256, 0, stream>>>(K, a, b, T, costp);
    cost_reduce<<<1, 256, 0, stream>>>(costp, out);
}
